// SelfAttention_28759101014369
// MI455X (gfx1250) — compile-verified, matrix operand order checked by reading, not run on hardware
//
#include <hip/hip_runtime.h>

typedef __bf16 bf16;
typedef bf16  v8bf  __attribute__((ext_vector_type(8)));
typedef bf16  v16bf __attribute__((ext_vector_type(16)));
typedef float v8f   __attribute__((ext_vector_type(8)));
typedef unsigned int u32x4 __attribute__((ext_vector_type(4)));
typedef int          i32x4 __attribute__((ext_vector_type(4)));
typedef int          i32x8 __attribute__((ext_vector_type(8)));

#define BATCH 4
#define SEQ   2048
#define DIM   1024

#define BM 128
#define BN 128
#define BK 32
#define LDSS 40   // bf16 elems per LDS row: 80 bytes (64B data + 16B pad), 16B aligned

#if defined(__HIP_DEVICE_COMPILE__) && __has_builtin(__builtin_amdgcn_tensor_load_to_lds) && __has_builtin(__builtin_amdgcn_s_wait_tensorcnt)
#define USE_TDM 1
#else
#define USE_TDM 0
#endif

// ---- WMMA fragment load from LDS (16-bit A/B layout per CDNA5 ISA 7.12.2) ----
// lanes m / m+16 hold row (row0 + m); K-halves 0..7 / 8..15 and 16..23 / 24..31.
static __device__ __forceinline__ v16bf lds_frag(const bf16* s, int row0, int lane) {
    const int r  = row0 + (lane & 15);
    const int kh = ((lane >> 4) & 1) * 8;
    const bf16* p = s + r * LDSS + kh;
    union { v16bf v; v8bf h[2]; } u;
    u.h[0] = *(const v8bf*)(p);        // K = kh .. kh+7
    u.h[1] = *(const v8bf*)(p + 16);   // K = kh+16 .. kh+23
    return u.v;
}

struct Acc { v8f c[4][2]; };   // wave tile 64(M) x 32(N)

#if USE_TDM
// Issue one TDM DMA: tile of `rows` x BK bf16 elements from global (row stride
// `ld` elements) into LDS at byte offset lds_addr, padding each 64B row with
// 16B so the LDS row stride is 80B (= LDSS bf16). ISA 08 / D# section 8.
static __device__ __forceinline__ void tdm_load_tile(
    unsigned lds_addr, const bf16* gptr, int ld, int rows)
{
    const unsigned long long ga = (unsigned long long)(size_t)gptr;
    u32x4 g0;
    g0[0] = 1u;                                   // count=1 (valid user descriptor)
    g0[1] = lds_addr;                             // lds_addr (bytes)
    g0[2] = (unsigned)(ga & 0xFFFFFFFFu);         // global_addr[31:0]
    g0[3] = (unsigned)((ga >> 32) & 0x01FFFFFFu)  // global_addr[56:32]
          | (2u << 30);                           // type = 2 ("image")

    const unsigned td0 = (unsigned)ld;            // tensor_dim0 (elements)
    const unsigned td1 = 1u << 24;                // tensor_dim1: large (no OOB)
    i32x8 g1;
    g1[0] = (int)((1u << 16)                      // data_size = 2 bytes
          |       (1u << 20)                      // pad_enable
          |       (3u << 22)                      // pad_interval: 16 DWORDs (64B)
          |       (3u << 25));                    // pad_amount: 4 DWORDs (16B)
    g1[1] = (int)((td0 & 0xFFFFu) << 16);         // [47:32]=abar addr 0, [63:48]=td0 lo
    g1[2] = (int)(((td0 >> 16) & 0xFFFFu)         // [79:64] = td0 hi
          |       ((td1 & 0xFFFFu) << 16));       // [95:80] = td1 lo
    g1[3] = (int)(((td1 >> 16) & 0xFFFFu)         // [111:96] = td1 hi
          |       ((unsigned)BK << 16));          // [127:112] = tile_dim0 = 32
    g1[4] = (int)((unsigned)rows);                // [143:128]=tile_dim1, [159:144]=tile_dim2=0
    g1[5] = (int)td0;                             // tensor_dim0_stride[31:0] (elements)
    g1[6] = 0;                                    // stride[47:32]=0, dim1_stride lo = 0
    g1[7] = 0;
    const i32x4 gz4 = {0, 0, 0, 0};
    const i32x8 gz8 = {0, 0, 0, 0, 0, 0, 0, 0};
    // amdgpu-toolchain (clang-23) 6-arg form: (g0, g1, g2, g3, g4, cpol)
    __builtin_amdgcn_tensor_load_to_lds(g0, g1, gz4, gz4, gz8, 0);
}
#else
// Fallback: cooperative global->VGPR->LDS staging (coalesced 16B chunks).
static __device__ __forceinline__ void stage_tiles(
    bf16* sA, bf16* sB,
    const bf16* __restrict__ A, const bf16* __restrict__ Bp,
    int lda, int ldb, int m0, int n0, int k0, int tid)
{
    #pragma unroll
    for (int it = 0; it < 2; ++it) {
        const int idx = it * 256 + tid;
        const int row = idx >> 2, col = (idx & 3) * 8;
        *(v8bf*)&sA[row * LDSS + col] =
            *(const v8bf*)&A[(size_t)(m0 + row) * lda + k0 + col];
    }
    #pragma unroll
    for (int it = 0; it < 2; ++it) {
        const int idx = it * 256 + tid;
        const int row = idx >> 2, col = (idx & 3) * 8;
        *(v8bf*)&sB[row * LDSS + col] =
            *(const v8bf*)&Bp[(size_t)(n0 + row) * ldb + k0 + col];
    }
}
#endif

// Block tile 128x128, 8 waves in 2(M) x 4(N) grid, wave tile 64x32 (4x2 WMMA accums).
// A: MxK row-major (ld = lda). Bp: B-transposed, i.e. NxK row-major (ld = ldb).
// Double-buffered LDS; wave 0 drives the Tensor Data Mover, one barrier per K-step.
static __device__ __forceinline__ void gemm_core(
    const bf16* __restrict__ A, const bf16* __restrict__ Bp,
    int lda, int ldb, int m0, int n0, int kEnd,
    bf16 (*sA)[BM * LDSS], bf16 (*sB)[BN * LDSS], Acc& acc)
{
    const int tid  = threadIdx.x;
    const int lane = tid & 31;
    const int wave = tid >> 5;
    const int wm   = (wave & 1) * 64;   // 2 waves along M
    const int wn   = (wave >> 1) * 32;  // 4 waves along N

    const v8f z = {0.f,0.f,0.f,0.f,0.f,0.f,0.f,0.f};
    #pragma unroll
    for (int i = 0; i < 4; ++i) { acc.c[i][0] = z; acc.c[i][1] = z; }

#if USE_TDM
    const unsigned ldsA[2] = { (unsigned)(size_t)&sA[0][0], (unsigned)(size_t)&sA[1][0] };
    const unsigned ldsB[2] = { (unsigned)(size_t)&sB[0][0], (unsigned)(size_t)&sB[1][0] };
    if (wave == 0) {
        tdm_load_tile(ldsA[0], A  + (size_t)m0 * lda, lda, BM);
        tdm_load_tile(ldsB[0], Bp + (size_t)n0 * ldb, ldb, BN);
        __builtin_amdgcn_s_wait_tensorcnt(0);
    }
    __syncthreads();
#else
    stage_tiles(sA[0], sB[0], A, Bp, lda, ldb, m0, n0, 0, tid);
    __syncthreads();
#endif

    int cur = 0;
    for (int k0 = 0; k0 < kEnd; k0 += BK) {
        // Kick DMA / staging of the next tile into the other buffer.
        if (k0 + BK < kEnd) {
#if USE_TDM
            if (wave == 0) {
                tdm_load_tile(ldsA[cur ^ 1], A  + (size_t)m0 * lda + (k0 + BK), lda, BM);
                tdm_load_tile(ldsB[cur ^ 1], Bp + (size_t)n0 * ldb + (k0 + BK), ldb, BN);
            }
#else
            stage_tiles(sA[cur ^ 1], sB[cur ^ 1], A, Bp, lda, ldb, m0, n0, k0 + BK, tid);
#endif
        }

        const bf16* cA = sA[cur];
        const bf16* cB = sB[cur];
        v16bf a[4], b[2];
        #pragma unroll
        for (int i = 0; i < 4; ++i) a[i] = lds_frag(cA, wm + 16 * i, lane);
        #pragma unroll
        for (int j = 0; j < 2; ++j) b[j] = lds_frag(cB, wn + 16 * j, lane);
        #pragma unroll
        for (int i = 0; i < 4; ++i)
            #pragma unroll
            for (int j = 0; j < 2; ++j)
                acc.c[i][j] = __builtin_amdgcn_wmma_f32_16x16x32_bf16(
                    false, a[i], false, b[j], (short)0, acc.c[i][j], false, false);

#if USE_TDM
        if (wave == 0) __builtin_amdgcn_s_wait_tensorcnt(0);  // next buffer landed
#endif
        __syncthreads();
        cur ^= 1;
    }
}

// ---------------- f32 -> bf16 convert ----------------
__global__ __launch_bounds__(256) void cvt_kernel(const float* __restrict__ in,
                                                  bf16* __restrict__ out, int n) {
    const int i = (blockIdx.x * 256 + threadIdx.x) * 4;
    if (i + 3 < n) {
        const float4 v = *(const float4*)(in + i);
        out[i]     = (bf16)v.x; out[i + 1] = (bf16)v.y;
        out[i + 2] = (bf16)v.z; out[i + 3] = (bf16)v.w;
    }
}

// ---------------- Q/K projection: Y = X @ W^T (bf16 out, optional scale) ----------------
__global__ __launch_bounds__(256) void proj_qk_kernel(const bf16* __restrict__ X,
                                                      const bf16* __restrict__ W,
                                                      bf16* __restrict__ Y, float scale) {
    __shared__ bf16 sA[2][BM * LDSS];
    __shared__ bf16 sB[2][BN * LDSS];
    const int m0 = blockIdx.x * BM;
    const int n0 = blockIdx.y * BN;
    Acc acc;
    gemm_core(X, W, DIM, DIM, m0, n0, DIM, sA, sB, acc);

    const int lane = threadIdx.x & 31;
    const int wave = threadIdx.x >> 5;
    const int wm = (wave & 1) * 64, wn = (wave >> 1) * 32;
    const int nc = lane & 15;
    const int rb = ((lane >> 4) & 1) * 8;
    #pragma unroll
    for (int i = 0; i < 4; ++i)
        #pragma unroll
        for (int j = 0; j < 2; ++j) {
            const int mBase = m0 + wm + 16 * i + rb;
            const int n     = n0 + wn + 16 * j + nc;
            #pragma unroll
            for (int r = 0; r < 8; ++r)
                Y[(size_t)(mBase + r) * DIM + n] = (bf16)(acc.c[i][j][r] * scale);
        }
}

// ---------------- V projection, stored transposed per batch: VT[b][d][s] ----------------
__global__ __launch_bounds__(256) void proj_v_kernel(const bf16* __restrict__ X,
                                                     const bf16* __restrict__ W,
                                                     bf16* __restrict__ VT) {
    __shared__ bf16 sA[2][BM * LDSS];
    __shared__ bf16 sB[2][BN * LDSS];
    const int m0 = blockIdx.x * BM;
    const int n0 = blockIdx.y * BN;
    Acc acc;
    gemm_core(X, W, DIM, DIM, m0, n0, DIM, sA, sB, acc);

    const int lane = threadIdx.x & 31;
    const int wave = threadIdx.x >> 5;
    const int wm = (wave & 1) * 64, wn = (wave >> 1) * 32;
    const int nc = lane & 15;
    const int rb = ((lane >> 4) & 1) * 8;
    #pragma unroll
    for (int i = 0; i < 4; ++i)
        #pragma unroll
        for (int j = 0; j < 2; ++j) {
            const int mBase = m0 + wm + 16 * i + rb;
            const int n     = n0 + wn + 16 * j + nc;
            #pragma unroll
            for (int r = 0; r < 8; ++r) {
                const int gm = mBase + r;
                const int b  = gm >> 11;        // / SEQ
                const int s  = gm & (SEQ - 1);
                VT[((size_t)b * DIM + n) * SEQ + s] = (bf16)acc.c[i][j][r];
            }
        }
}

// ---------------- scores = Q @ K^T, causal-masked, raw into attn region ----------------
__global__ __launch_bounds__(256) void scores_kernel(const bf16* __restrict__ Qh,
                                                     const bf16* __restrict__ Kh,
                                                     float* __restrict__ attn) {
    const int b  = blockIdx.z;
    const int m0 = blockIdx.x * BM;   // query block
    const int n0 = blockIdx.y * BN;   // key block
    if (n0 > m0 + (BM - 1)) return;   // fully masked block: skip (triangular dispatch)
    __shared__ bf16 sA[2][BM * LDSS];
    __shared__ bf16 sB[2][BN * LDSS];
    Acc acc;
    const size_t base = (size_t)b * SEQ * DIM;
    gemm_core(Qh + base, Kh + base, DIM, DIM, m0, n0, DIM, sA, sB, acc);

    float* out = attn + (size_t)b * SEQ * SEQ;
    const int lane = threadIdx.x & 31;
    const int wave = threadIdx.x >> 5;
    const int wm = (wave & 1) * 64, wn = (wave >> 1) * 32;
    const int nc = lane & 15;
    const int rb = ((lane >> 4) & 1) * 8;
    const float NEG_INF = -__builtin_inff();
    #pragma unroll
    for (int i = 0; i < 4; ++i)
        #pragma unroll
        for (int j = 0; j < 2; ++j) {
            const int qBase = m0 + wm + 16 * i + rb;
            const int k     = n0 + wn + 16 * j + nc;
            #pragma unroll
            for (int r = 0; r < 8; ++r) {
                const int q = qBase + r;
                out[(size_t)q * SEQ + k] = (k <= q) ? acc.c[i][j][r] : NEG_INF;
            }
        }
}

// ---------------- row softmax in place + bf16 P copy ----------------
__global__ __launch_bounds__(256) void softmax_kernel(float* __restrict__ attn,
                                                      bf16* __restrict__ Ph) {
    const int row = blockIdx.x;          // 0 .. B*SEQ-1
    const int q   = row & (SEQ - 1);
    float* sc = attn + (size_t)row * SEQ;
    bf16*  ph = Ph   + (size_t)row * SEQ;
    const int tid = threadIdx.x;
    __shared__ float red[256];

    float m = -__builtin_inff();
    for (int k = tid; k <= q; k += 256) m = fmaxf(m, sc[k]);
    red[tid] = m; __syncthreads();
    for (int s2 = 128; s2 > 0; s2 >>= 1) {
        if (tid < s2) red[tid] = fmaxf(red[tid], red[tid + s2]);
        __syncthreads();
    }
    const float rowMax = red[0];
    __syncthreads();

    float sum = 0.f;
    for (int k = tid; k <= q; k += 256) {
        const float e = __expf(sc[k] - rowMax);
        sc[k] = e; sum += e;
    }
    red[tid] = sum; __syncthreads();
    for (int s2 = 128; s2 > 0; s2 >>= 1) {
        if (tid < s2) red[tid] += red[tid + s2];
        __syncthreads();
    }
    const float inv = 1.0f / red[0];

    for (int k = tid; k <= q; k += 256) {
        const float p = sc[k] * inv;
        sc[k] = p; ph[k] = (bf16)p;
    }
    for (int k = q + 1 + tid; k < SEQ; k += 256) {
        sc[k] = 0.f; ph[k] = (bf16)0.f;
    }
}

// ---------------- O = P @ V (B' = V^T, K-contiguous); triangular K loop ----------------
__global__ __launch_bounds__(256) void av_kernel(const bf16* __restrict__ Ph,
                                                 const bf16* __restrict__ VT,
                                                 float* __restrict__ outp) {
    const int b  = blockIdx.z;
    const int m0 = blockIdx.x * BM;
    const int n0 = blockIdx.y * BN;
    __shared__ bf16 sA[2][BM * LDSS];
    __shared__ bf16 sB[2][BN * LDSS];
    Acc acc;
    gemm_core(Ph + (size_t)b * SEQ * SEQ, VT + (size_t)b * DIM * SEQ,
              SEQ, SEQ, m0, n0, m0 + BM /* causal: k <= q0+127 */, sA, sB, acc);

    float* o = outp + (size_t)b * SEQ * DIM;
    const int lane = threadIdx.x & 31;
    const int wave = threadIdx.x >> 5;
    const int wm = (wave & 1) * 64, wn = (wave >> 1) * 32;
    const int nc = lane & 15;
    const int rb = ((lane >> 4) & 1) * 8;
    #pragma unroll
    for (int i = 0; i < 4; ++i)
        #pragma unroll
        for (int j = 0; j < 2; ++j) {
            const int mBase = m0 + wm + 16 * i + rb;
            const int n     = n0 + wn + 16 * j + nc;
            #pragma unroll
            for (int r = 0; r < 8; ++r)
                o[(size_t)(mBase + r) * DIM + n] = acc.c[i][j][r];
        }
}

extern "C" void kernel_launch(void* const* d_in, const int* in_sizes, int n_in,
                              void* d_out, int out_size, void* d_ws, size_t ws_size,
                              hipStream_t stream) {
    (void)in_sizes; (void)n_in; (void)out_size; (void)ws_size;
    const float* x  = (const float*)d_in[0];
    // d_in[1] = mask (tril) — causality applied analytically
    const float* Wq = (const float*)d_in[2];
    const float* Wk = (const float*)d_in[3];
    const float* Wv = (const float*)d_in[4];

    char* p = (char*)d_ws;
    bf16* xh = (bf16*)p; p += (size_t)BATCH * SEQ * DIM * 2;
    bf16* wq = (bf16*)p; p += (size_t)DIM * DIM * 2;
    bf16* wk = (bf16*)p; p += (size_t)DIM * DIM * 2;
    bf16* wv = (bf16*)p; p += (size_t)DIM * DIM * 2;
    bf16* qh = (bf16*)p; p += (size_t)BATCH * SEQ * DIM * 2;
    bf16* kh = (bf16*)p; p += (size_t)BATCH * SEQ * DIM * 2;
    bf16* vt = (bf16*)p; p += (size_t)BATCH * DIM * SEQ * 2;
    bf16* ph = (bf16*)p; // BATCH*SEQ*SEQ*2

    float* outO    = (float*)d_out;
    float* outAttn = (float*)d_out + (size_t)BATCH * SEQ * DIM;

    const int nX = BATCH * SEQ * DIM;   // 8388608
    const int nW = DIM * DIM;           // 1048576
    cvt_kernel<<<nX / 1024, 256, 0, stream>>>(x,  xh, nX);
    cvt_kernel<<<nW / 1024, 256, 0, stream>>>(Wq, wq, nW);
    cvt_kernel<<<nW / 1024, 256, 0, stream>>>(Wk, wk, nW);
    cvt_kernel<<<nW / 1024, 256, 0, stream>>>(Wv, wv, nW);

    const dim3 gProj(BATCH * SEQ / BM, DIM / BN);         // 64 x 8
    proj_qk_kernel<<<gProj, 256, 0, stream>>>(xh, wq, qh, 0.03125f); // 1/sqrt(1024)
    proj_qk_kernel<<<gProj, 256, 0, stream>>>(xh, wk, kh, 1.0f);
    proj_v_kernel <<<gProj, 256, 0, stream>>>(xh, wv, vt);

    scores_kernel<<<dim3(SEQ / BM, SEQ / BN, BATCH), 256, 0, stream>>>(qh, kh, outAttn);
    softmax_kernel<<<BATCH * SEQ, 256, 0, stream>>>(outAttn, ph);
    av_kernel<<<dim3(SEQ / BM, DIM / BN, BATCH), 256, 0, stream>>>(ph, vt, outO);
}
